// ParametricSTRF_32710470926517
// MI455X (gfx1250) — matrix operand order repaired and checked by reading, not compiled, not run on hardware
//
#include <hip/hip_runtime.h>

// ---------------- problem constants ----------------
#define FDIM 49      // frequency extent (kernel == input -> output F dim = 1)
#define TDIM 25      // time taps
#define OCH  16      // output channels == WMMA M
#define GRP  4
#define TIN  8000
#define BATCH 64

#define TT   128     // time columns per workgroup (8 waves x 16 cols)
#define HALO 24      // causal halo
#define XW   160     // padded LDS row width (need 159: HALO + TT + 7 pad-tau slack)

#define AFRAG_U32 (FDIM * 2 * 32 * 8)   // f x {hi,lo} x lane x 8 dwords = 25088 u32

typedef __attribute__((ext_vector_type(16))) __bf16 v16bf;
typedef __attribute__((ext_vector_type(8)))  float  v8f;
typedef __attribute__((ext_vector_type(4)))  int    v4i;

typedef __attribute__((address_space(1))) v4i* gv4i_p;   // global int4*
typedef __attribute__((address_space(3))) v4i* lv4i_p;   // LDS int4*

union Frag { unsigned u[8]; v16bf v; };

__device__ __forceinline__ unsigned short f2bf(float x) {
    unsigned u = __float_as_uint(x);
    unsigned r = ((u >> 16) & 1u) + 0x7FFFu;      // round-to-nearest-even
    return (unsigned short)((u + r) >> 16);
}
__device__ __forceinline__ float bf2f(unsigned short h) {
    return __uint_as_float(((unsigned)h) << 16);
}

// one-instruction byte gather: result bytes selected from {b,a} by sel
__device__ __forceinline__ unsigned perm32(unsigned hi_src, unsigned lo_src,
                                           unsigned sel) {
#if __has_builtin(__builtin_amdgcn_perm)
    return __builtin_amdgcn_perm(hi_src, lo_src, sel);
#else
    unsigned r = 0;
    for (int i = 0; i < 4; ++i) {
        unsigned s = (sel >> (8 * i)) & 0xFF;
        unsigned byte = (s < 4) ? (lo_src >> (8 * s)) : (hi_src >> (8 * (s - 4)));
        r |= (byte & 0xFF) << (8 * i);
    }
    return r;
#endif
}

// -------------------------------------------------------------------------
// Kernel 1: build DCLS gaussian kernel, emit pre-swizzled WMMA A-fragments
// (hi/lo bf16 planes) into global scratch.
// A-fragment dword mapping (16-bit A 16x32, contiguous K order used here, same as B):
//   lane L (row M = L%16), dword j holds K = kbase, kbase+1 where
//   kbase = 2j + (L<16 ? 0 : 16)
// grid = 49*2 blocks (f, level), 256 threads = 32 lanes x 8 dwords.
// -------------------------------------------------------------------------
__global__ __launch_bounds__(256) void build_afrag(
        const float* __restrict__ W, const float* __restrict__ P,
        const float* __restrict__ SIG, unsigned* __restrict__ afrag)
{
    const int f    = blockIdx.x >> 1;
    const int lvl  = blockIdx.x & 1;
    const int tid  = threadIdx.x;
    const int lane = tid >> 3;
    const int j    = tid & 7;
    const int o    = lane & 15;
    const int kbase = 2 * j + ((lane < 16) ? 0 : 16);

    unsigned short val[2];
    #pragma unroll
    for (int d = 0; d < 2; ++d) {
        const int tau = kbase + d;
        float acc = 0.0f;
        if (tau < TDIM) {
            for (int g = 0; g < GRP; ++g) {
                const float muf = P[o * GRP + g] + (float)(FDIM / 2);
                const float mut = P[64 + o * GRP + g] + (float)(TDIM / 2);
                const float sf  = fabsf(SIG[o * GRP + g]) + 0.27f;
                const float st  = fabsf(SIG[64 + o * GRP + g]) + 0.27f;
                float sumf = 0.0f, sumt = 0.0f;
                for (int ff = 0; ff < FDIM; ++ff) {
                    float dd = ((float)ff - muf) / sf; sumf += expf(-0.5f * dd * dd);
                }
                for (int t2 = 0; t2 < TDIM; ++t2) {
                    float dd = ((float)t2 - mut) / st; sumt += expf(-0.5f * dd * dd);
                }
                const float df = ((float)f - muf) / sf;
                const float dt = ((float)tau - mut) / st;
                const float xv = expf(-0.5f * (df * df + dt * dt));
                acc += W[o * GRP + g] * xv / (sumf * sumt + 1e-7f);
            }
        }
        const unsigned short h = f2bf(acc);
        val[d] = (lvl == 0) ? h : f2bf(acc - bf2f(h));
    }
    afrag[((f * 2 + lvl) * 32 + lane) * 8 + j] =
        (unsigned)val[0] | ((unsigned)val[1] << 16);
}

// -------------------------------------------------------------------------
// Kernel 2: the conv as 16xN GEMM on bf16 WMMA with hi/lo (bf16x3) split.
// Each wave: one 16(chan) x 16(time) f32 tile; 49 f-steps, each one K=32
// (tau zero-padded) slab -> 3x v_wmma_f32_16x16x32_bf16 into 3 independent
// accumulators (no intra-step WMMA->WMMA RAW chain).
// -------------------------------------------------------------------------
__global__ __launch_bounds__(256) void dcls_main(
        const float* __restrict__ x, const unsigned* __restrict__ afrag_g,
        float* __restrict__ out)
{
    __shared__ unsigned sA[AFRAG_U32];   // 100,352 B  pre-swizzled A frags
    __shared__ unsigned sX[FDIM * XW];   //  31,360 B  packed {hi,lo} bf16 x tile

    const int tid  = threadIdx.x;
    const int lane = tid & 31;
    const int wave = tid >> 5;
    const int b    = blockIdx.y;
    const int T0   = blockIdx.x * TT;

    // ---- stage A fragments global -> LDS via async-to-LDS DMA ----
#if __has_builtin(__builtin_amdgcn_global_load_async_to_lds_b128) && \
    __has_builtin(__builtin_amdgcn_s_wait_asynccnt)
    for (int c4 = tid; c4 < AFRAG_U32 / 4; c4 += 256) {
        __builtin_amdgcn_global_load_async_to_lds_b128(
            (gv4i_p)(afrag_g + c4 * 4),
            (lv4i_p)(&sA[c4 * 4]),
            0, 0);
    }
    __builtin_amdgcn_s_wait_asynccnt(0);
#else
    for (int c = tid; c < AFRAG_U32; c += 256) sA[c] = afrag_g[c];
#endif

    // ---- stage x tile: f32 -> {hi,lo} bf16 packed u32 ----
    const float* xb = x + (size_t)b * FDIM * TIN;
    for (int idx = tid; idx < FDIM * XW; idx += 256) {
        const int f  = idx / XW;
        const int tt = idx - f * XW;
        const int t  = T0 - HALO + tt;
        const float v = (t >= 0 && t < TIN) ? xb[f * TIN + t] : 0.0f;
        const unsigned short h = f2bf(v);
        const unsigned short l = f2bf(v - bf2f(h));
        sX[idx] = (unsigned)h | ((unsigned)l << 16);
    }
    __syncthreads();

    // ---- per-wave output tile ----
    const int ct    = blockIdx.x * 8 + wave;   // column-tile id (16 cols each)
    int       t0    = ct * 16;
    const bool valid = (t0 < TIN);
    if (!valid) t0 = T0;                       // clamp: keep LDS reads in-bounds

    const int n      = lane & 15;              // B column within tile
    const int kbaseB = (lane < 16) ? 0 : 16;   // B K-range (tau offset)
    const int ti0    = (t0 - T0) + n + kbaseB; // local time index of K=kbaseB

    v8f acc_hh = {};    // hi(A) * hi(B)
    v8f acc_hl = {};    // hi(A) * lo(B)
    v8f acc_lh = {};    // lo(A) * hi(B)
    for (int f = 0; f < FDIM; ++f) {
        // A fragments (hi, lo): contiguous 32B per lane
        Frag ahi, alo;
        const unsigned* pa = &sA[((f * 2 + 0) * 32 + lane) * 8];
        #pragma unroll
        for (int j = 0; j < 8; ++j) { ahi.u[j] = pa[j]; alo.u[j] = pa[256 + j]; }

        // B fragments: 16 consecutive packed time samples -> hi & lo frags
        const unsigned* px = &sX[f * XW + ti0];
        unsigned r[16];
        #pragma unroll
        for (int m = 0; m < 16; ++m) r[m] = px[m];

        Frag bhi, blo;
        #pragma unroll
        for (int j = 0; j < 8; ++j) {
            const unsigned r0 = r[2 * j], r1 = r[2 * j + 1];
            bhi.u[j] = perm32(r1, r0, 0x05040100u);  // low  halves {r0.lo, r1.lo}
            blo.u[j] = perm32(r1, r0, 0x07060302u);  // high halves {r0.hi, r1.hi}
        }

        // bf16x3: three independent accumulation chains
        acc_hh = __builtin_amdgcn_wmma_f32_16x16x32_bf16(false, ahi.v, false, bhi.v,
                                                         (short)0, acc_hh, false, false);
        acc_hl = __builtin_amdgcn_wmma_f32_16x16x32_bf16(false, ahi.v, false, blo.v,
                                                         (short)0, acc_hl, false, false);
        acc_lh = __builtin_amdgcn_wmma_f32_16x16x32_bf16(false, alo.v, false, bhi.v,
                                                         (short)0, acc_lh, false, false);
    }

    // ---- reduce + store: C/D layout -> out[b, o, 0, t0+n] ----
    if (valid) {
        const int o0 = (lane < 16) ? 0 : 8;
        float* ob = out + (size_t)b * OCH * TIN + t0 + n;
        #pragma unroll
        for (int r8 = 0; r8 < 8; ++r8) {
            ob[(size_t)(r8 + o0) * TIN] = acc_hh[r8] + (acc_hl[r8] + acc_lh[r8]);
        }
    }
}

extern "C" void kernel_launch(void* const* d_in, const int* in_sizes, int n_in,
                              void* d_out, int out_size, void* d_ws, size_t ws_size,
                              hipStream_t stream) {
    const float* x   = (const float*)d_in[0];
    const float* W   = (const float*)d_in[1];
    const float* P   = (const float*)d_in[2];
    const float* SIG = (const float*)d_in[3];
    float*    out    = (float*)d_out;
    unsigned* afrag  = (unsigned*)d_ws;

    hipLaunchKernelGGL(build_afrag, dim3(FDIM * 2), dim3(256), 0, stream,
                       W, P, SIG, afrag);

    // 8000 / 16 = 500 column tiles per batch; 8 tiles per WG -> 63 WGs
    dim3 grid((TIN / 16 + 7) / 8, BATCH);
    hipLaunchKernelGGL(dcls_main, grid, dim3(256), 0, stream, x, afrag, out);
}
